// MPNNEncoder_33749853012259
// MI455X (gfx1250) — matrix operand order repaired and checked by reading, not compiled, not run on hardware
//
#include <hip/hip_runtime.h>

// ---------------------------------------------------------------- constants
#define E_EDGES   102400
#define NN_NODES  51200
#define NM_MOLS   2048
#define HID       300
#define NPAD      320          // HIDDEN padded to multiple of 32
#define NT_TILES  (NPAD / 16)  // 20 N-tiles
#define EF_REAL   147
#define EF_PAD    160
#define NF_REAL   133
#define AIK_PAD   448          // (133 + 300) padded
#define MAX_NB    6

#define BLK_M     512          // per workgroup (8 waves x 64 rows)
#define BLK_N     32           // per workgroup (2 wmma tiles)

typedef __attribute__((ext_vector_type(16))) __bf16 v16bf;
typedef __attribute__((ext_vector_type(8)))  float  v8f;
typedef unsigned uint4v __attribute__((ext_vector_type(4)));

struct FragBF { union { v16bf v; unsigned u[8]; uint4v q[2]; }; };

// ---------------------------------------------------------------- bf16 utils
__device__ __forceinline__ unsigned short f2bf(float x) {
  unsigned u = __builtin_bit_cast(unsigned, x);
  u += 0x7FFFu + ((u >> 16) & 1u);            // round-to-nearest-even
  return (unsigned short)(u >> 16);
}
__device__ __forceinline__ float bflo(unsigned u) { return __builtin_bit_cast(float, u << 16); }
__device__ __forceinline__ float bfhi(unsigned u) { return __builtin_bit_cast(float, u & 0xFFFF0000u); }
__device__ __forceinline__ float bf2f(unsigned short s) { return __builtin_bit_cast(float, ((unsigned)s) << 16); }

__device__ __forceinline__ unsigned bfsub1(unsigned x, unsigned y) {
  float lo = bflo(x) - bflo(y);
  float hi = bfhi(x) - bfhi(y);
  return (unsigned)f2bf(lo) | ((unsigned)f2bf(hi) << 16);
}
__device__ __forceinline__ uint4v bfsub4(uint4v x, uint4v y) {
  uint4v r;
  r[0] = bfsub1(x[0], y[0]); r[1] = bfsub1(x[1], y[1]);
  r[2] = bfsub1(x[2], y[2]); r[3] = bfsub1(x[3], y[3]);
  return r;
}

#define WMMA_BF16(A, B, C) \
  __builtin_amdgcn_wmma_f32_16x16x32_bf16(false, (A), false, (B), (short)0, (C), false, false)

// ---------------------------------------------------------------- packing
// weights: f32 [Kreal x Nreal] row-major -> fragment-major WMMA B layout:
//   dst[((kt*NT_TILES + nt)*32 + lane)*8 + j] = {bf16 B[k,n], bf16 B[k+1,n]}
//   with n = nt*16 + (lane&15), k = kt*32 + (lane>>4)*8 + (j>>2)*16 + 2*(j&3)
__global__ __launch_bounds__(256) void k_pack_bfrag(const float* __restrict__ src,
                                                    unsigned* __restrict__ dst,
                                                    int Kreal, int Nreal, int Ktiles) {
  int idx = blockIdx.x * 256 + threadIdx.x;
  if (idx >= Ktiles * NT_TILES * 32 * 8) return;
  int j    = idx & 7;
  int lane = (idx >> 3) & 31;
  int rest = idx >> 8;
  int nt   = rest % NT_TILES;
  int kt   = rest / NT_TILES;
  int n    = nt * 16 + (lane & 15);
  int k0   = kt * 32 + (lane >> 4) * 8 + (j >> 2) * 16 + 2 * (j & 3);
  float a = (k0     < Kreal && n < Nreal) ? src[(size_t)k0       * Nreal + n] : 0.f;
  float b = (k0 + 1 < Kreal && n < Nreal) ? src[(size_t)(k0 + 1) * Nreal + n] : 0.f;
  dst[idx] = (unsigned)f2bf(a) | ((unsigned)f2bf(b) << 16);
}

// f32 [rows x Nreal] -> bf16 [rows x Np] zero-padded
__global__ __launch_bounds__(256) void k_pack_rows(const float* __restrict__ src,
                                                   unsigned short* __restrict__ dst,
                                                   int Nreal, int Np, long total) {
  long idx = (long)blockIdx.x * 256 + threadIdx.x;
  if (idx >= total) return;
  long r = idx / Np; int c = (int)(idx % Np);
  dst[idx] = (c < Nreal) ? f2bf(src[(size_t)r * Nreal + c]) : (unsigned short)0;
}

// B fragment pair load: 2x b128, contiguous per lane
__device__ __forceinline__ void load_bfrag2(const unsigned* __restrict__ Bf, int kt, int nt0,
                                            int lane, FragBF& b0, FragBF& b1) {
  const uint4v* bp = (const uint4v*)(Bf + (((size_t)kt * NT_TILES + nt0) * 32 + lane) * 8);
  b0.q[0] = bp[0];  b0.q[1] = bp[1];
  b1.q[0] = bp[16]; b1.q[1] = bp[17];   // +16 b128 = +64 u32 = 8 lanes * 8 u32 (an N-tile is 256 u32)
}

// ---------------------------------------------------------------- GEMM: inp = f_edges @ W_i ; message = relu(inp)
__global__ __launch_bounds__(256) void k_gemm_wi(const unsigned short* __restrict__ A,
                                                 const unsigned* __restrict__ Bf,
                                                 unsigned short* __restrict__ inp,
                                                 unsigned short* __restrict__ msg) {
  const int lane = threadIdx.x & 31, wave = threadIdx.x >> 5;
  const int row0  = blockIdx.y * BLK_M + wave * 64;
  const int ncol0 = blockIdx.x * BLK_N;
  const int nt0   = blockIdx.x * 2;
  const int lm = lane & 15, kh = lane >> 4;

  v8f acc[4][2];
  v8f zero = {0.f,0.f,0.f,0.f,0.f,0.f,0.f,0.f};
#pragma unroll
  for (int i = 0; i < 4; ++i) { acc[i][0] = zero; acc[i][1] = zero; }

  const unsigned short* arow[4];
#pragma unroll
  for (int i = 0; i < 4; ++i)
    arow[i] = A + (size_t)(row0 + i * 16 + lm) * EF_PAD + kh * 8;

  for (int kc = 0; kc < EF_PAD; kc += 32) {
    FragBF b0, b1;
    load_bfrag2(Bf, kc >> 5, nt0, lane, b0, b1);
#pragma unroll
    for (int i = 0; i < 4; ++i) {
      FragBF a;
      const uint4v* p = (const uint4v*)(arow[i] + kc);
      a.q[0] = p[0]; a.q[1] = p[2];
      acc[i][0] = WMMA_BF16(a.v, b0.v, acc[i][0]);
      acc[i][1] = WMMA_BF16(a.v, b1.v, acc[i][1]);
    }
  }
#pragma unroll
  for (int i = 0; i < 4; ++i)
#pragma unroll
    for (int t = 0; t < 2; ++t)
#pragma unroll
      for (int r = 0; r < 8; ++r) {
        int row = row0 + i * 16 + r + kh * 8;
        int col = ncol0 + t * 16 + lm;
        size_t idx = (size_t)row * NPAD + col;
        float v = acc[i][t][r];
        inp[idx] = f2bf(v);
        msg[idx] = f2bf(v > 0.f ? v : 0.f);
      }
}

// ---------------------------------------------------------------- GEMM: msgOut = relu(inp + (nmsg[e2n]-msg[e2rev]) @ W_h)
__global__ __launch_bounds__(256) void k_gemm_wh(const unsigned short* __restrict__ nmsg,
                                                 const unsigned short* __restrict__ msgIn,
                                                 const int* __restrict__ e2n,
                                                 const int* __restrict__ e2rev,
                                                 const unsigned* __restrict__ Bf,
                                                 const unsigned short* __restrict__ inp,
                                                 unsigned short* __restrict__ msgOut) {
  const int lane = threadIdx.x & 31, wave = threadIdx.x >> 5;
  const int row0  = blockIdx.y * BLK_M + wave * 64;
  const int ncol0 = blockIdx.x * BLK_N;
  const int nt0   = blockIdx.x * 2;
  const int lm = lane & 15, kh = lane >> 4;

  v8f acc[4][2];
  v8f zero = {0.f,0.f,0.f,0.f,0.f,0.f,0.f,0.f};
#pragma unroll
  for (int i = 0; i < 4; ++i) { acc[i][0] = zero; acc[i][1] = zero; }

  const unsigned short *prow[4], *qrow[4];
#pragma unroll
  for (int i = 0; i < 4; ++i) {
    int e = row0 + i * 16 + lm;
    prow[i] = nmsg  + (size_t)e2n[e]   * NPAD + kh * 8;
    qrow[i] = msgIn + (size_t)e2rev[e] * NPAD + kh * 8;
  }

  for (int kc = 0; kc < NPAD; kc += 32) {
    FragBF b0, b1;
    load_bfrag2(Bf, kc >> 5, nt0, lane, b0, b1);
#pragma unroll
    for (int i = 0; i < 4; ++i) {
      const uint4v* pp = (const uint4v*)(prow[i] + kc);
      const uint4v* qq = (const uint4v*)(qrow[i] + kc);
      FragBF a;
      a.q[0] = bfsub4(pp[0], qq[0]);
      a.q[1] = bfsub4(pp[2], qq[2]);
      acc[i][0] = WMMA_BF16(a.v, b0.v, acc[i][0]);
      acc[i][1] = WMMA_BF16(a.v, b1.v, acc[i][1]);
    }
  }
#pragma unroll
  for (int i = 0; i < 4; ++i)
#pragma unroll
    for (int t = 0; t < 2; ++t)
#pragma unroll
      for (int r = 0; r < 8; ++r) {
        int row = row0 + i * 16 + r + kh * 8;
        int col = ncol0 + t * 16 + lm;
        size_t idx = (size_t)row * NPAD + col;
        float v = bf2f(inp[idx]) + acc[i][t][r];
        msgOut[idx] = f2bf(v > 0.f ? v : 0.f);
      }
}

// ---------------------------------------------------------------- GEMM: hidden = relu(a_input @ W_o + b_o)  (f32 out, stride NPAD)
__global__ __launch_bounds__(256) void k_gemm_wo(const unsigned short* __restrict__ A,
                                                 const unsigned* __restrict__ Bf,
                                                 const float* __restrict__ b_o,
                                                 float* __restrict__ hidden) {
  const int lane = threadIdx.x & 31, wave = threadIdx.x >> 5;
  const int row0  = blockIdx.y * BLK_M + wave * 64;
  const int ncol0 = blockIdx.x * BLK_N;
  const int nt0   = blockIdx.x * 2;
  const int lm = lane & 15, kh = lane >> 4;

  v8f acc[4][2];
  v8f zero = {0.f,0.f,0.f,0.f,0.f,0.f,0.f,0.f};
#pragma unroll
  for (int i = 0; i < 4; ++i) { acc[i][0] = zero; acc[i][1] = zero; }

  const unsigned short* arow[4];
#pragma unroll
  for (int i = 0; i < 4; ++i)
    arow[i] = A + (size_t)(row0 + i * 16 + lm) * AIK_PAD + kh * 8;

  for (int kc = 0; kc < AIK_PAD; kc += 32) {
    FragBF b0, b1;
    load_bfrag2(Bf, kc >> 5, nt0, lane, b0, b1);
#pragma unroll
    for (int i = 0; i < 4; ++i) {
      FragBF a;
      const uint4v* p = (const uint4v*)(arow[i] + kc);
      a.q[0] = p[0]; a.q[1] = p[2];
      acc[i][0] = WMMA_BF16(a.v, b0.v, acc[i][0]);
      acc[i][1] = WMMA_BF16(a.v, b1.v, acc[i][1]);
    }
  }
#pragma unroll
  for (int i = 0; i < 4; ++i)
#pragma unroll
    for (int t = 0; t < 2; ++t) {
      int col = ncol0 + t * 16 + lm;
      float bo = (col < HID) ? b_o[col] : 0.f;
#pragma unroll
      for (int r = 0; r < 8; ++r) {
        int row = row0 + i * 16 + r + kh * 8;
        float v = acc[i][t][r] + bo;
        hidden[(size_t)row * NPAD + col] = v > 0.f ? v : 0.f;
      }
    }
}

// ---------------------------------------------------------------- neighbor aggregation: nmsg[v] = sum_k msg[n2e[v][k]]
__global__ __launch_bounds__(256) void k_aggregate(const unsigned short* __restrict__ msg,
                                                   const int* __restrict__ n2e,
                                                   unsigned short* __restrict__ nmsg) {
  int idx = blockIdx.x * 256 + threadIdx.x;
  if (idx >= NN_NODES * (NPAD / 8)) return;
  int v = idx / (NPAD / 8);
  int g = idx % (NPAD / 8);
  const int* nb = n2e + v * MAX_NB;
  float s[8];
#pragma unroll
  for (int j = 0; j < 8; ++j) s[j] = 0.f;
#pragma unroll
  for (int k = 0; k < MAX_NB; ++k) {
    uint4v q = *(const uint4v*)(msg + (size_t)nb[k] * NPAD + g * 8);
#pragma unroll
    for (int j = 0; j < 4; ++j) { s[2*j] += bflo(q[j]); s[2*j+1] += bfhi(q[j]); }
  }
  uint4v o;
#pragma unroll
  for (int j = 0; j < 4; ++j)
    o[j] = (unsigned)f2bf(s[2*j]) | ((unsigned)f2bf(s[2*j+1]) << 16);
  *(uint4v*)(nmsg + (size_t)idx * 8) = o;
}

// ---------------------------------------------------------------- a_input = concat(f_nodes, n_msg) bf16, K padded to 448
__global__ __launch_bounds__(256) void k_build_ainput(const float* __restrict__ f_nodes,
                                                      const unsigned short* __restrict__ nmsg,
                                                      unsigned short* __restrict__ ainp) {
  long idx = (long)blockIdx.x * 256 + threadIdx.x;
  if (idx >= (long)NN_NODES * AIK_PAD) return;
  int v = (int)(idx / AIK_PAD);
  int c = (int)(idx % AIK_PAD);
  unsigned short val;
  if (c < NF_REAL)            val = f2bf(f_nodes[(size_t)v * NF_REAL + c]);
  else if (c < NF_REAL + HID) val = nmsg[(size_t)v * NPAD + (c - NF_REAL)];
  else                        val = 0;
  ainp[idx] = val;
}

// ---------------------------------------------------------------- segment mean (mol_ids sorted -> binary search, deterministic)
struct SegB { int s, e; };

__global__ __launch_bounds__(256) void k_seg_bounds(const int* __restrict__ mol_ids,
                                                    SegB* __restrict__ bounds) {
  int m = blockIdx.x * 256 + threadIdx.x;
  if (m >= NM_MOLS) return;
  int lo = 0, hi = NN_NODES;
  while (lo < hi) { int mid = (lo + hi) >> 1; if (mol_ids[mid] < m) lo = mid + 1; else hi = mid; }
  int s = lo;
  hi = NN_NODES;
  while (lo < hi) { int mid = (lo + hi) >> 1; if (mol_ids[mid] < m + 1) lo = mid + 1; else hi = mid; }
  bounds[m].s = s; bounds[m].e = lo;
}

__global__ __launch_bounds__(256) void k_seg_mean(const float* __restrict__ hidden,
                                                  const SegB* __restrict__ bounds,
                                                  float* __restrict__ out) {
  int idx = blockIdx.x * 256 + threadIdx.x;
  if (idx >= NM_MOLS * HID) return;
  int m = idx / HID, c = idx % HID;
  SegB b = bounds[m];
  int cnt = b.e - b.s;
  float s = 0.f;
  for (int v = b.s; v < b.e; ++v) s += hidden[(size_t)v * NPAD + c];
  out[idx] = (cnt > 0) ? s / (float)cnt : 0.f;
}

// ---------------------------------------------------------------- launch
extern "C" void kernel_launch(void* const* d_in, const int* in_sizes, int n_in,
                              void* d_out, int out_size, void* d_ws, size_t ws_size,
                              hipStream_t stream) {
  const float* f_nodes = (const float*)d_in[0];
  const float* f_edges = (const float*)d_in[1];
  const float* W_i     = (const float*)d_in[2];
  const float* W_h     = (const float*)d_in[3];
  const float* W_o     = (const float*)d_in[4];
  const float* b_o     = (const float*)d_in[5];
  const int*   n2e     = (const int*)d_in[6];
  const int*   e2n     = (const int*)d_in[7];
  const int*   e2rev   = (const int*)d_in[8];
  const int*   mol_ids = (const int*)d_in[9];
  float* out = (float*)d_out;

  char* p = (char*)d_ws;
  auto alloc = [&](size_t bytes) { void* r = (void*)p; p += (bytes + 255) & ~(size_t)255; return r; };

  unsigned short* fe_p  = (unsigned short*)alloc((size_t)E_EDGES * EF_PAD * 2);
  unsigned short* inp   = (unsigned short*)alloc((size_t)E_EDGES * NPAD * 2);
  unsigned short* msgA  = (unsigned short*)alloc((size_t)E_EDGES * NPAD * 2);
  unsigned short* msgB  = (unsigned short*)alloc((size_t)E_EDGES * NPAD * 2);
  unsigned short* nmsg  = (unsigned short*)alloc((size_t)NN_NODES * NPAD * 2);
  unsigned*       Wi_f  = (unsigned*)alloc((size_t)(EF_PAD / 32) * NT_TILES * 32 * 8 * 4);
  unsigned*       Wh_f  = (unsigned*)alloc((size_t)(NPAD / 32) * NT_TILES * 32 * 8 * 4);
  unsigned*       Wo_f  = (unsigned*)alloc((size_t)(AIK_PAD / 32) * NT_TILES * 32 * 8 * 4);
  unsigned short* ainp  = (unsigned short*)alloc((size_t)NN_NODES * AIK_PAD * 2);
  float*          hid   = (float*)alloc((size_t)NN_NODES * NPAD * 4);
  SegB*           bnd   = (SegB*)alloc((size_t)NM_MOLS * sizeof(SegB));

  // pack weights into fragment-major WMMA B layout (zero padded)
  k_pack_bfrag<<<((EF_PAD / 32) * NT_TILES * 256 + 255) / 256, 256, 0, stream>>>(W_i, Wi_f, EF_REAL, HID, EF_PAD / 32);
  k_pack_bfrag<<<((NPAD / 32) * NT_TILES * 256 + 255) / 256, 256, 0, stream>>>(W_h, Wh_f, HID, HID, NPAD / 32);
  k_pack_bfrag<<<((AIK_PAD / 32) * NT_TILES * 256 + 255) / 256, 256, 0, stream>>>(W_o, Wo_f, NF_REAL + HID, HID, AIK_PAD / 32);

  // pack edge features to bf16 (zero padded)
  {
    long tot = (long)E_EDGES * EF_PAD;
    k_pack_rows<<<(unsigned)((tot + 255) / 256), 256, 0, stream>>>(f_edges, fe_p, EF_REAL, EF_PAD, tot);
  }

  // inp = f_edges @ W_i ; message = relu(inp)
  dim3 gE(NPAD / BLK_N, E_EDGES / BLK_M);
  k_gemm_wi<<<gE, 256, 0, stream>>>(fe_p, Wi_f, inp, msgA);

  unsigned short* cur = msgA;
  unsigned short* nxt = msgB;
  int aggBlocks = NN_NODES * (NPAD / 8) / 256;
  for (int it = 0; it < 5; ++it) {   // DEPTH - 1
    k_aggregate<<<aggBlocks, 256, 0, stream>>>(cur, n2e, nmsg);
    k_gemm_wh<<<gE, 256, 0, stream>>>(nmsg, cur, e2n, e2rev, Wh_f, inp, nxt);
    unsigned short* t = cur; cur = nxt; nxt = t;
  }

  // final aggregation + readout
  k_aggregate<<<aggBlocks, 256, 0, stream>>>(cur, n2e, nmsg);
  {
    long tot = (long)NN_NODES * AIK_PAD;
    k_build_ainput<<<(unsigned)((tot + 255) / 256), 256, 0, stream>>>(f_nodes, nmsg, ainp);
  }
  dim3 gN(NPAD / BLK_N, NN_NODES / BLK_M);
  k_gemm_wo<<<gN, 256, 0, stream>>>(ainp, Wo_f, b_o, hid);

  // segment mean
  k_seg_bounds<<<(NM_MOLS + 255) / 256, 256, 0, stream>>>(mol_ids, bnd);
  k_seg_mean<<<(NM_MOLS * HID + 255) / 256, 256, 0, stream>>>(hid, bnd, out);
}
